// NaLP_69767448756638
// MI455X (gfx1250) — matrix operand order repaired and checked by reading, not compiled, not run on hardware
//
#include <hip/hip_runtime.h>
#include <hip/hip_bf16.h>

// ---------------------------------------------------------------------------
// NaLP forward, fused for MI455X (gfx1250, wave32, WMMA + async LDS copy).
// B=4096, ARITY=6, EMB=100, NF=200, FC=1200.
//
// ws layout (bytes):
//   m      f32 [24576 x 200]        @ 0          (19,660,800)
//   scale  f32 [200]                @ 19,660,800
//   shift  f32 [200]                @ 19,661,600
//   mrel   f16 [24576 x 224]        @ 19,662,400 (11,010,048)  (K padded 200->224)
//   gwh    f16 [1200 x 448]         @ 30,672,448 ( 1,075,200)  (K padded 400->448)
// ---------------------------------------------------------------------------

typedef __attribute__((ext_vector_type(16))) _Float16 v16h;
typedef __attribute__((ext_vector_type(8)))  _Float16 v8h;
typedef __attribute__((ext_vector_type(8)))  float    v8f;
typedef __attribute__((ext_vector_type(4)))  int      v4i;

union AU { v8h h[2]; v16h v; };

#define NROWS 24576      // B * ARITY
#define NF    200
#define KPAD  224        // padded per-half K
#define KP2   448        // padded pair K (14 * 32)
#define FC    1200
#define BB    4          // batches per block in the big kernel

// gfx1250 async global->LDS path (guarded: falls back to sync copy if the
// toolchain doesn't declare the builtins).  Probe round 2 confirmed the
// builtin exists with param type "int4 addrspace(1)*" (prints as __device__).
#if defined(__AMDGCN__) && \
    __has_builtin(__builtin_amdgcn_global_load_async_to_lds_b128) && \
    __has_builtin(__builtin_amdgcn_s_wait_asynccnt)
#define USE_ASYNC_LDS 1
typedef v4i __attribute__((address_space(1)))* gv4ip;   // global int4*
typedef v4i __attribute__((address_space(3)))* lv4ip;   // LDS int4*
#else
#define USE_ASYNC_LDS 0
#endif

// ---------------------------------------------------------------------------
// Kernel 1: embedding gather + conv GEMM (fp32).  One block per (b, a) row.
// ---------------------------------------------------------------------------
__global__ __launch_bounds__(256) void k_gather_conv(
    const int* __restrict__ x, const float* __restrict__ er,
    const float* __restrict__ ev, const float* __restrict__ cw,
    const float* __restrict__ cb, float* __restrict__ m)
{
  __shared__ float cat[2 * 100];
  const int row = blockIdx.x;            // b*6 + a
  const int b = row / 6, a = row % 6;
  const int t = threadIdx.x;
  const int ri = x[b * 12 + 2 * a];
  const int vi = x[b * 12 + 2 * a + 1];
  if (t < 200)
    cat[t] = (t < 100) ? er[(size_t)ri * 100 + t]
                       : ev[(size_t)vi * 100 + (t - 100)];
  __syncthreads();
  if (t < 200) {
    const float* w = cw + (size_t)t * 200;
    float acc = cb[t];
#pragma unroll 4
    for (int k = 0; k < 200; ++k) acc += cat[k] * w[k];
    m[(size_t)row * NF + t] = acc;
  }
}

// ---------------------------------------------------------------------------
// Kernel 2: BN statistics per feature; folds gamma/beta into scale/shift.
// ---------------------------------------------------------------------------
__global__ __launch_bounds__(256) void k_bnstats(
    const float* __restrict__ m, const float* __restrict__ gamma,
    const float* __restrict__ beta, float* __restrict__ scale,
    float* __restrict__ shiftv)
{
  __shared__ float s1[256], s2[256];
  const int f = blockIdx.x;
  const int t = threadIdx.x;
  float sum = 0.f, sq = 0.f;
  for (int r = t; r < NROWS; r += 256) {
    float v = m[(size_t)r * NF + f];
    sum += v; sq += v * v;
  }
  s1[t] = sum; s2[t] = sq;
  __syncthreads();
  for (int s = 128; s > 0; s >>= 1) {
    if (t < s) { s1[t] += s1[t + s]; s2[t] += s2[t + s]; }
    __syncthreads();
  }
  if (t == 0) {
    const float inv = 1.f / (float)NROWS;
    float mu  = s1[0] * inv;
    float var = s2[0] * inv - mu * mu;
    float rstd = rsqrtf(var + 1e-5f);
    float sc = gamma[f] * rstd;
    scale[f]  = sc;
    shiftv[f] = beta[f] - mu * sc;
  }
}

// ---------------------------------------------------------------------------
// Kernel 3: BN apply + ReLU + convert to f16, pad feature dim 200 -> 224.
// ---------------------------------------------------------------------------
__global__ __launch_bounds__(256) void k_bnrelu_h(
    const float* __restrict__ m, const float* __restrict__ scale,
    const float* __restrict__ shiftv, _Float16* __restrict__ mrel)
{
  const int i = blockIdx.x * 256 + threadIdx.x;
  if (i >= NROWS * KPAD) return;
  const int row = i / KPAD, c = i % KPAD;
  float v = 0.f;
  if (c < NF) {
    v = m[(size_t)row * NF + c] * scale[c] + shiftv[c];
    v = v > 0.f ? v : 0.f;
  }
  mrel[i] = (_Float16)v;
}

// ---------------------------------------------------------------------------
// Kernel 3b: repack gW (1200 x 400 f32) -> gwh (1200 x 448 f16), padded.
// ---------------------------------------------------------------------------
__global__ __launch_bounds__(256) void k_pack_gw(
    const float* __restrict__ gW, _Float16* __restrict__ gwh)
{
  const int i = blockIdx.x * 256 + threadIdx.x;
  if (i >= FC * KP2) return;
  const int o = i / KP2, c = i % KP2;
  float v = 0.f;
  if (c < NF)                          v = gW[(size_t)o * 400 + c];
  else if (c >= KPAD && c < KPAD + NF) v = gW[(size_t)o * 400 + (c - 24)];
  gwh[i] = (_Float16)v;
}

// ---------------------------------------------------------------------------
// Kernel 4: fused pair-GEMM (WMMA f16->f32) + bias + ReLU + min over 36 pairs
//           + final 1200-dot with fW.  One block = BB=4 batches = 144 rows
//           = 9 M-tiles of 16.  N = 1200 = 75 tiles of 16.  K = 448 = 14x32.
// B tiles are double-buffered in LDS; on gfx1250 they are staged with
// GLOBAL_LOAD_ASYNC_TO_LDS_B128 overlapping the previous tile's WMMA work.
// ---------------------------------------------------------------------------
__global__ __launch_bounds__(256) void k_pair_gemm_min(
    const _Float16* __restrict__ mrel,   // (24576, 224)
    const _Float16* __restrict__ gwh,    // (1200, 448)
    const float* __restrict__ gb,        // 1200
    const float* __restrict__ fW,        // 1200
    const float* __restrict__ fb,        // 1
    float* __restrict__ out)             // 4096
{
  __shared__ __align__(64) _Float16 aLDS[BB * 6 * KPAD];   // 10.5 KB
  __shared__ __align__(64) _Float16 bLDS[2][16 * KP2];     // 2 x 14.0 KB
  __shared__ int   minbuf[BB * FC];                        // 18.75 KB
  __shared__ float red[256];

  const int tid  = threadIdx.x;
  const int lane = tid & 31;
  const int wave = tid >> 5;
  const int bb0  = blockIdx.x * BB;

  // min-accumulators start at +inf (bit pattern valid: post-ReLU values >= 0)
  for (int i = tid; i < BB * FC; i += 256) minbuf[i] = 0x7f800000;

  // ---- staging helpers -----------------------------------------------------
  auto stageB = [&](int nt, int buf) {
    const _Float16* src = gwh + (size_t)nt * 16 * KP2;
#if USE_ASYNC_LDS
    for (int i = tid; i < (16 * KP2) / 8; i += 256)
      __builtin_amdgcn_global_load_async_to_lds_b128(
          (gv4ip)(src + i * 8), (lv4ip)&bLDS[buf][i * 8], 0, 0);
#else
    uint4* dst = (uint4*)&bLDS[buf][0];
    const uint4* s = (const uint4*)src;
    for (int i = tid; i < (16 * KP2) / 8; i += 256) dst[i] = s[i];
#endif
  };

  // stage the BB*6 activation vectors into LDS
  {
    const _Float16* src = mrel + (size_t)bb0 * 6 * KPAD;
#if USE_ASYNC_LDS
    for (int i = tid; i < (BB * 6 * KPAD) / 8; i += 256)
      __builtin_amdgcn_global_load_async_to_lds_b128(
          (gv4ip)(src + i * 8), (lv4ip)&aLDS[i * 8], 0, 0);
#else
    const uint4* s = (const uint4*)src;
    uint4* dst = (uint4*)aLDS;
    for (int i = tid; i < (BB * 6 * KPAD) / 8; i += 256) dst[i] = s[i];
#endif
  }

  // per-lane constants for the WMMA 16-bit operands (contiguous K order k = 16h + s)
  const int mrow  = lane & 15;               // A row within tile
  const int grp   = (lane < 16) ? 0 : 16;    // A: K half offset
  const int koffB = (lane < 16) ? 0 : 16;    // B: K half offset
  const int ncol  = lane & 15;               // B/D column within tile

  // ---- one N-tile of compute ----------------------------------------------
  auto compute = [&](int nt, int buf) {
    const int   o   = nt * 16 + ncol;
    const float gbv = gb[o];

    for (int mt = wave; mt < (BB * 36) / 16; mt += 8) {   // 9 M-tiles, 8 waves
      const int r  = mt * 16 + mrow;        // row within block [0,144)
      const int bl = r / 36;                // local batch
      const int p  = r % 36;                // pair index
      const int vfirst  = bl * 6 + (p % 6); // first 224 K -> m_rel[b, p%6]
      const int vsecond = bl * 6 + (p / 6); // last  224 K -> m_rel[b, p/6]

      v8f acc = {};
#pragma unroll
      for (int ks = 0; ks < KP2 / 32; ++ks) {
        const int k0 = ks * 32;
        const int vsel = (k0 < KPAD) ? vfirst : vsecond;
        const int kcol = (k0 < KPAD) ? k0 : (k0 - KPAD);
        // A fragment: two contiguous 8-half (16B) chunks per lane
        const v8h* ap = (const v8h*)&aLDS[vsel * KPAD + kcol + grp];
        AU au;
        au.h[0] = ap[0];          // K = k0+grp+0..7
        au.h[1] = ap[1];          // K = k0+grp+8..15
        // B fragment: 16 contiguous halfs per lane (col = ncol)
        v16h bfrag = *(const v16h*)&bLDS[buf][ncol * KP2 + k0 + koffB];
        acc = __builtin_amdgcn_wmma_f32_16x16x32_f16(
            false, au.v, false, bfrag, (short)0, acc, false, false);
      }

      // epilogue: + gb, ReLU, min over pairs via LDS integer atomic-min
#pragma unroll
      for (int i = 0; i < 8; ++i) {
        float v = acc[i] + gbv;
        v = v > 0.f ? v : 0.f;
        const int rr  = mt * 16 + i + ((lane < 16) ? 0 : 8);  // D-row layout
        const int bl2 = rr / 36;
        atomicMin(&minbuf[bl2 * FC + o], __float_as_int(v));
      }
    }
  };

  // ---- main loop ------------------------------------------------------------
#if USE_ASYNC_LDS
  stageB(0, 0);                              // overlaps nothing, but async
  __builtin_amdgcn_s_wait_asynccnt(0);       // aLDS + bLDS[0] resident
  __syncthreads();
  for (int nt = 0; nt < FC / 16; ++nt) {
    if (nt + 1 < FC / 16) {
      stageB(nt + 1, (nt + 1) & 1);          // async fill of other buffer
      __builtin_prefetch(gwh + (size_t)(nt + 1) * 16 * KP2 + tid * 28, 0, 0);
    }
    compute(nt, nt & 1);
    __builtin_amdgcn_s_wait_asynccnt(0);     // my async stores to LDS done
    __syncthreads();                         // everyone's done; buffers swap
  }
#else
  for (int nt = 0; nt < FC / 16; ++nt) {
    __syncthreads();                         // readers of bLDS[0] finished
    stageB(nt, 0);
    if (nt + 1 < FC / 16)
      __builtin_prefetch(gwh + (size_t)(nt + 1) * 16 * KP2 + tid * 28, 0, 0);
    __syncthreads();                         // writers finished
    compute(nt, 0);
  }
  __syncthreads();
#endif

  // score[b] = fb + sum_o min_val[b,o] * fW[o]
  for (int bl = 0; bl < BB; ++bl) {
    float partial = 0.f;
    for (int o2 = tid; o2 < FC; o2 += 256)
      partial += __int_as_float(minbuf[bl * FC + o2]) * fW[o2];
    red[tid] = partial;
    __syncthreads();
    for (int s = 128; s > 0; s >>= 1) {
      if (tid < s) red[tid] += red[tid + s];
      __syncthreads();
    }
    if (tid == 0) out[bb0 + bl] = red[0] + fb[0];
    __syncthreads();
  }
}

// ---------------------------------------------------------------------------
extern "C" void kernel_launch(void* const* d_in, const int* in_sizes, int n_in,
                              void* d_out, int out_size, void* d_ws, size_t ws_size,
                              hipStream_t stream) {
  const int*   x      = (const int*)  d_in[0];   // (4096, 12) indices
  const float* er     = (const float*)d_in[1];   // (5000, 100)
  const float* ev     = (const float*)d_in[2];   // (50000, 100)
  const float* cw     = (const float*)d_in[3];   // (200, 200)
  const float* cb     = (const float*)d_in[4];   // (200,)
  const float* gamma  = (const float*)d_in[5];   // (200,)
  const float* beta   = (const float*)d_in[6];   // (200,)
  const float* gW     = (const float*)d_in[7];   // (1200, 400)
  const float* gb     = (const float*)d_in[8];   // (1200,)
  const float* fW     = (const float*)d_in[9];   // (1, 1200)
  const float* fb     = (const float*)d_in[10];  // (1,)
  float* out = (float*)d_out;                    // (4096, 1)

  char* ws = (char*)d_ws;
  float*     m      = (float*)     (ws + 0);
  float*     scale  = (float*)     (ws + 19660800);
  float*     shiftv = (float*)     (ws + 19661600);
  _Float16*  mrel   = (_Float16*)  (ws + 19662400);
  _Float16*  gwh    = (_Float16*)  (ws + 30672448);

  k_gather_conv<<<NROWS, 256, 0, stream>>>(x, er, ev, cw, cb, m);
  k_bnstats    <<<NF,    256, 0, stream>>>(m, gamma, beta, scale, shiftv);
  k_bnrelu_h   <<<(NROWS * KPAD + 255) / 256, 256, 0, stream>>>(m, scale, shiftv, mrel);
  k_pack_gw    <<<(FC * KP2 + 255) / 256, 256, 0, stream>>>(gW, gwh);
  k_pair_gemm_min<<<4096 / BB, 256, 0, stream>>>(mrel, gwh, gb, fW, fb, out);
}
